// TridentNet_77189152243864
// MI455X (gfx1250) — matrix operand order repaired and checked by reading, not compiled, not run on hardware
//
#include <hip/hip_runtime.h>

typedef unsigned short ushort_t;
typedef __attribute__((ext_vector_type(16))) __bf16 v16bf;
typedef __attribute__((ext_vector_type(8)))  __bf16 v8bf;
typedef __attribute__((ext_vector_type(8)))  float  v8f;
typedef __attribute__((ext_vector_type(4)))  unsigned int uint4v;

#define EPSV 1e-5f

__device__ __forceinline__ ushort_t f2bf(float f) {
    union { float f; unsigned u; } v; v.f = f;
    unsigned r = v.u + 0x7FFFu + ((v.u >> 16) & 1u);   // round-to-nearest-even
    return (ushort_t)(r >> 16);
}
__device__ __forceinline__ float bf2f(ushort_t h) {
    union { unsigned u; float f; } v; v.u = ((unsigned)h) << 16;
    return v.f;
}

// CDNA5 async global->LDS copy (ASYNCcnt-tracked, bypasses VGPRs).
// VDST = wave-relative LDS byte offset, VADDR = 64-bit global address.
__device__ __forceinline__ void async_ld_b128(unsigned lds_off, const void* gaddr) {
    asm volatile("global_load_async_to_lds_b128 %0, %1, off"
                 :: "v"(lds_off), "v"(gaddr)
                 : "memory");
}
__device__ __forceinline__ void wait_async0() {
    asm volatile("s_wait_asynccnt 0" ::: "memory");
}

// ---------------------------------------------------------------------------
// WMMA GEMM: Y[M,Nc](bf16) = A[M,Kpad](bf16) @ W[K,Nc], weights pre-packed as
// Wt[Nc][Kpad] (N-major, bf16, zero-padded K). M%256==0, Kpad%32==0, Nc%64==0.
// 256 threads = 8 waves; WG tile 256x64, wave tile 32x64 (8 WMMA / K-step).
// Double-buffered LDS filled by async global->LDS b128 copies.
// ---------------------------------------------------------------------------
__global__ __launch_bounds__(256) void gemm_bf16(
    const ushort_t* __restrict__ A, const ushort_t* __restrict__ Wt,
    ushort_t* __restrict__ Y, int M, int Kpad, int Nc)
{
    __shared__ ushort_t sA[2][256][32];   // 2 x 16 KB
    __shared__ ushort_t sB[2][64][32];    // 2 x  4 KB
    const int tid  = threadIdx.x;
    const int wave = tid >> 5;
    const int lane = tid & 31;
    const int m0 = blockIdx.x * 256;
    const int n0 = blockIdx.y * 64;
    const int ra = tid >> 2;              // 0..63
    const int co = (tid & 3) * 8;         // 0/8/16/24 (bf16 elements)

    v8f acc[2][4];
    #pragma unroll
    for (int s = 0; s < 2; ++s)
        #pragma unroll
        for (int nf = 0; nf < 4; ++nf)
            #pragma unroll
            for (int t = 0; t < 8; ++t) acc[s][nf][t] = 0.0f;

    auto issue = [&](int buf, int kt) {
        #pragma unroll
        for (int q = 0; q < 4; ++q) {      // A tile: 256 rows x 32 cols
            int r = ra + q * 64;
            async_ld_b128((unsigned)(size_t)&sA[buf][r][co],
                          &A[(size_t)(m0 + r) * Kpad + kt + co]);
        }
        async_ld_b128((unsigned)(size_t)&sB[buf][ra][co],   // B tile: 64 x 32
                      &Wt[(size_t)(n0 + ra) * Kpad + kt + co]);
    };

    const int nk = Kpad >> 5;
    issue(0, 0);
    for (int ki = 0; ki < nk; ++ki) {
        const int buf = ki & 1;
        wait_async0();            // my async copies for this tile done
        __syncthreads();          // everyone's copies visible
        if (ki + 1 < nk) issue(buf ^ 1, (ki + 1) << 5);

        // A fragments (16x32 each): lane<16 holds K 0..7 & 16..23 of its row,
        // lane>=16 holds K 8..15 & 24..31.
        const int mr = wave * 32 + (lane & 15);
        const int kb = (lane < 16) ? 0 : 8;
        v16bf a0, a1;
        ((v8bf*)&a0)[0] = *(const v8bf*)&sA[buf][mr][kb];
        ((v8bf*)&a0)[1] = *(const v8bf*)&sA[buf][mr][kb + 16];
        ((v8bf*)&a1)[0] = *(const v8bf*)&sA[buf][mr + 16][kb];
        ((v8bf*)&a1)[1] = *(const v8bf*)&sA[buf][mr + 16][kb + 16];

        const int kb2 = (lane < 16) ? 0 : 8;
        #pragma unroll
        for (int nf = 0; nf < 4; ++nf) {
            const int ncol = nf * 16 + (lane & 15);
            v16bf b;
            ((v8bf*)&b)[0] = *(const v8bf*)&sB[buf][ncol][kb2];
            ((v8bf*)&b)[1] = *(const v8bf*)&sB[buf][ncol][kb2 + 16];
            acc[0][nf] = __builtin_amdgcn_wmma_f32_16x16x32_bf16(
                false, a0, false, b, (short)0, acc[0][nf], false, false);
            acc[1][nf] = __builtin_amdgcn_wmma_f32_16x16x32_bf16(
                false, a1, false, b, (short)0, acc[1][nf], false, false);
        }
    }

    // C/D layout: VGPR r holds M = r (lanes 0-15) / 8+r (lanes 16-31), N = lane&15
    const int mo  = (lane < 16) ? 0 : 8;
    const int ncl = lane & 15;
    #pragma unroll
    for (int s = 0; s < 2; ++s) {
        const int mrb = m0 + wave * 32 + s * 16 + mo;
        #pragma unroll
        for (int nf = 0; nf < 4; ++nf)
            #pragma unroll
            for (int r = 0; r < 8; ++r)
                Y[(size_t)(mrb + r) * Nc + n0 + nf * 16 + ncl] = f2bf(acc[s][nf][r]);
    }
}

// ---------------------------------------------------------------------------
// BN batch statistics: one block per channel, deterministic LDS tree reduce.
// Reads bf16 (Yb) or f32 (Yf). mv[c]=mean, mv[C+c]=rsqrt(var+eps).
// ---------------------------------------------------------------------------
__global__ __launch_bounds__(256) void bn_stats(
    const ushort_t* Yb, const float* Yf, long long Mtot, int stride, float* mv)
{
    __shared__ float s1[256], s2[256];
    const int c = blockIdx.x, t = threadIdx.x;
    float sum = 0.f, sq = 0.f;
    for (long long i = t; i < Mtot; i += 256) {
        float v = Yb ? bf2f(Yb[i * (long long)stride + c])
                     : Yf[i * (long long)stride + c];
        sum += v; sq += v * v;
    }
    s1[t] = sum; s2[t] = sq; __syncthreads();
    for (int s = 128; s > 0; s >>= 1) {
        if (t < s) { s1[t] += s1[t + s]; s2[t] += s2[t + s]; }
        __syncthreads();
    }
    if (t == 0) {
        float m = s1[0] / (float)Mtot;
        float var = s2[0] / (float)Mtot - m * m;
        mv[c] = m;
        mv[gridDim.x + c] = rsqrtf(var + EPSV);
    }
}

// fused BN + ReLU, requantize to bf16
__global__ __launch_bounds__(256) void bn_relu(
    const ushort_t* Y, ushort_t* out, const float* mv,
    const float* g, const float* b, long long tot, int C)
{
    long long gid = (long long)blockIdx.x * 256 + threadIdx.x;
    if (gid >= tot) return;
    int c = (int)(gid % C);
    float v = (bf2f(Y[gid]) - mv[c]) * mv[C + c] * g[c] + b[c];
    out[gid] = f2bf(fmaxf(v, 0.f));
}

// input BN apply -> bf16 features with zero-padded stride
__global__ __launch_bounds__(256) void bn_apply_pad(
    const float* x, const float* mv, const float* g, const float* b,
    ushort_t* fts, int C, int stride, long long rows)
{
    long long gid = (long long)blockIdx.x * 256 + threadIdx.x;
    if (gid >= rows * stride) return;
    long long r = gid / stride; int c = (int)(gid % stride);
    ushort_t o = 0;
    if (c < C) {
        float v = (x[r * C + c] - mv[c]) * mv[C + c] * g[c] + b[c];
        o = f2bf(v);
    }
    fts[gid] = o;
}

// pos f32 [P,3] -> bf16 [P,4] zero-padded
__global__ __launch_bounds__(256) void prep_pos(const float* pos, ushort_t* pb, long long P)
{
    long long gid = (long long)blockIdx.x * 256 + threadIdx.x;
    if (gid >= P * 4) return;
    long long r = gid >> 2; int c = (int)(gid & 3);
    pb[gid] = (c < 3) ? f2bf(pos[r * 3 + c]) : (ushort_t)0;
}

// squared norms per point
__global__ __launch_bounds__(256) void sqnorm(
    const ushort_t* pts, int stride, int D, float* sq, long long P)
{
    long long gid = (long long)blockIdx.x * 256 + threadIdx.x;
    if (gid >= P) return;
    float s = 0.f;
    for (int d = 0; d < D; ++d) { float v = bf2f(pts[gid * stride + d]); s += v * v; }
    sq[gid] = s;
}

// ---------------------------------------------------------------------------
// kNN (k=16): per-batch 300x300 distances, LDS-tiled j points, register top-16.
// grid = (ceil(N/128), B), block = 128
// ---------------------------------------------------------------------------
__global__ __launch_bounds__(128) void knn_kernel(
    const ushort_t* pts, const float* sq, int* idx, int N, int D, int stride)
{
    __shared__ ushort_t tile[64 * 128];
    const int b = blockIdx.y;
    const int i = blockIdx.x * 128 + threadIdx.x;
    const ushort_t* base = pts + (size_t)b * N * stride;
    const float* sqb = sq + (size_t)b * N;

    float di[16]; int ii[16];
    #pragma unroll
    for (int q = 0; q < 16; ++q) { di[q] = 3.4e38f; ii[q] = 0; }
    const float sqi = (i < N) ? sqb[i] : 0.f;

    for (int j0 = 0; j0 < N; j0 += 64) {
        int cnt = min(64, N - j0);
        for (int t = threadIdx.x; t < cnt * D; t += 128) {
            int jr = t / D, d = t - jr * D;
            tile[jr * D + d] = base[(size_t)(j0 + jr) * stride + d];
        }
        __syncthreads();
        if (i < N) {
            for (int jj = 0; jj < cnt; jj += 8) {
                float dot[8];
                #pragma unroll
                for (int q = 0; q < 8; ++q) dot[q] = 0.f;
                for (int d = 0; d < D; ++d) {
                    float pid = bf2f(base[(size_t)i * stride + d]);
                    #pragma unroll
                    for (int q = 0; q < 8; ++q)
                        dot[q] += pid * bf2f(tile[(jj + q) * D + d]);
                }
                int jb = min(8, cnt - jj);
                for (int q = 0; q < jb; ++q) {
                    int j = j0 + jj + q;
                    if (j == i) continue;
                    float dj = sqi + sqb[j] - 2.f * dot[q];
                    if (dj < di[15]) {
                        di[15] = dj; ii[15] = j;
                        #pragma unroll
                        for (int s = 15; s > 0; --s) {
                            if (di[s] < di[s - 1]) {
                                float td = di[s]; di[s] = di[s - 1]; di[s - 1] = td;
                                int ti = ii[s]; ii[s] = ii[s - 1]; ii[s - 1] = ti;
                            }
                        }
                    }
                }
            }
        }
        __syncthreads();
    }
    if (i < N) {
        #pragma unroll
        for (int q = 0; q < 16; ++q) idx[((size_t)b * N + i) * 16 + q] = ii[q];
    }
}

// ---------------------------------------------------------------------------
// Edge feature build: H[r, :] = [u_b, x_i, x_j - x_i, 0-pad], bf16, 8 cols/thread
// ---------------------------------------------------------------------------
__global__ __launch_bounds__(256) void edge_build(
    const ushort_t* u, int U, int Ust,
    const ushort_t* fts, int C, int Fst,
    const int* idx, ushort_t* H, int Kpad, int N, long long total)
{
    long long gid = (long long)blockIdx.x * 256 + threadIdx.x;
    if (gid >= total) return;
    const int cpr = Kpad >> 3;
    long long r = gid / cpr;
    int col0 = (int)(gid % cpr) * 8;
    long long bn = r >> 4;               // k==16
    int j = (int)(r & 15);
    long long b = bn / N;
    long long n = bn % N;
    int jidx = idx[bn * 16 + j];
    const ushort_t* xi = fts + (b * N + n) * Fst;
    const ushort_t* xj = fts + (b * N + jidx) * Fst;
    const ushort_t* ub = u + b * Ust;
    ushort_t out[8];
    #pragma unroll
    for (int q = 0; q < 8; ++q) {
        int col = col0 + q;
        ushort_t v = 0;
        if (col < U)               v = ub[col];
        else if (col < U + C)      v = xi[col - U];
        else if (col < U + 2 * C)  v = f2bf(bf2f(xj[col - U - C]) - bf2f(xi[col - U - C]));
        out[q] = v;
    }
    *(uint4v*)&H[r * Kpad + col0] = *(uint4v*)out;
}

// max over k edges + BN'd skip + ReLU -> next node features (bf16)
__global__ __launch_bounds__(256) void aggr_skip(
    const ushort_t* H3, const ushort_t* Ys, const float* mvs,
    const float* gs, const float* bs, ushort_t* fo, long long tot, int C)
{
    long long gid = (long long)blockIdx.x * 256 + threadIdx.x;
    if (gid >= tot) return;
    long long n = gid / C; int c = (int)(gid % C);
    float m = -3.4e38f;
    #pragma unroll
    for (int j = 0; j < 16; ++j)
        m = fmaxf(m, bf2f(H3[(n * 16 + j) * C + c]));
    float s = (bf2f(Ys[gid]) - mvs[c]) * mvs[C + c] * gs[c] + bs[c];
    fo[gid] = f2bf(fmaxf(m + s, 0.f));
}

// per-batch mean pool over nodes
__global__ __launch_bounds__(256) void meanpool(
    const ushort_t* fts, int stride, int C, int N, float* out, long long tot)
{
    long long gid = (long long)blockIdx.x * 256 + threadIdx.x;
    if (gid >= tot) return;
    long long b = gid / C; int c = (int)(gid % C);
    float s = 0.f;
    for (int n = 0; n < N; ++n) s += bf2f(fts[(b * N + n) * stride + c]);
    out[gid] = s / (float)N;
}

// small dense f32 linear: out[b,o] = [in0|in1] @ W + bias, optional ReLU
__global__ __launch_bounds__(256) void linear2(
    const float* in0, int K0, const float* in1, int K1,
    const float* W, const float* bias, float* out, int Bn, int O, int doRelu)
{
    int gid = blockIdx.x * 256 + threadIdx.x;
    if (gid >= Bn * O) return;
    int b = gid / O, o = gid % O;
    float s = bias ? bias[o] : 0.f;
    for (int i = 0; i < K0; ++i) s += in0[(size_t)b * K0 + i] * W[(size_t)i * O + o];
    for (int i = 0; i < K1; ++i) s += in1[(size_t)b * K1 + i] * W[(size_t)(K0 + i) * O + o];
    if (doRelu) s = fmaxf(s, 0.f);
    out[gid] = s;
}

// BN over batch axis (B=128) + ReLU; writes f32 and bf16 copies of u
__global__ __launch_bounds__(128) void bn_cols_relu(
    const float* y, const float* g, const float* be,
    float* uo, ushort_t* ub, int Bn, int O)
{
    __shared__ float s1[128], s2[128];
    const int o = blockIdx.x, t = threadIdx.x;
    float v = y[(size_t)t * O + o];
    s1[t] = v; s2[t] = v * v; __syncthreads();
    for (int s = 64; s > 0; s >>= 1) {
        if (t < s) { s1[t] += s1[t + s]; s2[t] += s2[t + s]; }
        __syncthreads();
    }
    float m = s1[0] / (float)Bn;
    float var = s2[0] / (float)Bn - m * m;
    float r = rsqrtf(var + EPSV);
    float out = fmaxf((v - m) * r * g[o] + be[o], 0.f);
    uo[(size_t)t * O + o] = out;
    ub[(size_t)t * O + o] = f2bf(out);
}

// weight pack: f32 W[K,Nc] -> bf16 Wt[Nc][Kpad] (N-major, K zero-padded)
__global__ __launch_bounds__(256) void wpack(
    const float* W, ushort_t* Wt, int K, int Nc, int Kpad)
{
    int gid = blockIdx.x * 256 + threadIdx.x;
    if (gid >= Nc * Kpad) return;
    int n = gid / Kpad, kk = gid % Kpad;
    Wt[gid] = (kk < K) ? f2bf(W[(size_t)kk * Nc + n]) : (ushort_t)0;
}

// ===========================================================================
extern "C" void kernel_launch(void* const* d_in, const int* in_sizes, int n_in,
                              void* d_out, int out_size, void* d_ws, size_t ws_size,
                              hipStream_t stream)
{
    (void)in_sizes; (void)n_in; (void)out_size; (void)ws_size;
    const int B = 128, N = 300;
    const long long BN = (long long)B * N;        // 38400 (=150*256)
    const long long M  = BN * 16;                 // 614400 (=2400*256)

    // ---- inputs (setup_inputs dict order, flattened) ----
    const float* pos   = (const float*)d_in[0];
    const float* xin   = (const float*)d_in[1];
    const float* ibn_g = (const float*)d_in[2];
    const float* ibn_b = (const float*)d_in[3];
    // gp[i] at 4+4i: W, bias, gamma, beta
    // conv[i] at 20+12i: W1,g1,b1,W2,g2,b2,W3,g3,b3,Ws,gs,bs
    const float* fcW  = (const float*)d_in[56];
    const float* fcb  = (const float*)d_in[57];
    const float* outW = (const float*)d_in[58];
    const float* outb = (const float*)d_in[59];

    // ---- workspace carve-out ----
    char* ws = (char*)d_ws; size_t off = 0;
    auto alloc = [&](size_t bytes) -> void* {
        void* p = ws + off; off += (bytes + 255) & ~(size_t)255; return p;
    };
    ushort_t* bufA  = (ushort_t*)alloc((size_t)M * 384 * 2);   // H / activations
    ushort_t* bufY  = (ushort_t*)alloc((size_t)M * 256 * 2);   // pre-BN GEMM out
    ushort_t* ftsA  = (ushort_t*)alloc((size_t)BN * 256 * 2);
    ushort_t* ftsB  = (ushort_t*)alloc((size_t)BN * 256 * 2);
    ushort_t* skipY = (ushort_t*)alloc((size_t)BN * 256 * 2);
    ushort_t* posb  = (ushort_t*)alloc((size_t)BN * 4 * 2);
    float*    sqn   = (float*)alloc((size_t)BN * 4);
    int*      idx   = (int*)alloc((size_t)BN * 16 * 4);
    ushort_t* wt0   = (ushort_t*)alloc((size_t)384 * 256 * 2);
    ushort_t* wt1   = (ushort_t*)alloc((size_t)256 * 256 * 2);
    float* mvIn  = (float*)alloc(2 * 16 * 4);
    float* mvE   = (float*)alloc(2 * 256 * 4);
    float* mvS   = (float*)alloc(2 * 256 * 4);
    float* pool  = (float*)alloc((size_t)B * 256 * 4);
    float* gy    = (float*)alloc((size_t)B * 256 * 4);
    float* u_f   = (float*)alloc((size_t)B * 256 * 4);
    ushort_t* u_b = (ushort_t*)alloc((size_t)B * 256 * 2);
    float* hbuf  = (float*)alloc((size_t)B * 256 * 4);

    auto cdiv = [](long long a, long long b) { return (int)((a + b - 1) / b); };

    // ---- 1. input BN over (B*N) + padded bf16 features (stride 32) ----
    bn_stats<<<16, 256, 0, stream>>>(nullptr, xin, BN, 16, mvIn);
    bn_apply_pad<<<cdiv(BN * 32, 256), 256, 0, stream>>>(
        xin, mvIn, ibn_g, ibn_b, ftsA, 16, 32, BN);

    // ---- 2. u0 = relu(bn(meanpool @ W0 + b0)) ----
    meanpool<<<cdiv((long long)B * 16, 256), 256, 0, stream>>>(
        ftsA, 32, 16, N, pool, (long long)B * 16);
    linear2<<<cdiv((long long)B * 16, 256), 256, 0, stream>>>(
        pool, 16, nullptr, 0, (const float*)d_in[4], (const float*)d_in[5],
        gy, B, 16, 0);
    bn_cols_relu<<<16, 128, 0, stream>>>(
        gy, (const float*)d_in[6], (const float*)d_in[7], u_f, u_b, B, 16);

    prep_pos<<<cdiv(BN * 4, 256), 256, 0, stream>>>(pos, posb, BN);

    // ---- 3. three DynamicEdgeConv layers ----
    const int Cin[3]  = {16, 64, 128};
    const int Cout[3] = {64, 128, 256};
    const int Fst[3]  = {32, 64, 128};    // bf16 row stride of fts (K-padded)
    const int Kp1[3]  = {64, 192, 384};   // padded first-MLP K (= 3*Cin rounded to 32)

    ushort_t* ftsCur = ftsA;
    for (int ci = 0; ci < 3; ++ci) {
        const int C = Cin[ci], O = Cout[ci], U = Cin[ci];
        ushort_t* ftsNext = (ftsCur == ftsA) ? ftsB : ftsA;
        const int pbase = 20 + ci * 12;
        const float* W1 = (const float*)d_in[pbase + 0];
        const float* g1 = (const float*)d_in[pbase + 1];
        const float* b1 = (const float*)d_in[pbase + 2];
        const float* W2 = (const float*)d_in[pbase + 3];
        const float* g2 = (const float*)d_in[pbase + 4];
        const float* b2 = (const float*)d_in[pbase + 5];
        const float* W3 = (const float*)d_in[pbase + 6];
        const float* g3 = (const float*)d_in[pbase + 7];
        const float* b3 = (const float*)d_in[pbase + 8];
        const float* Wsk = (const float*)d_in[pbase + 9];
        const float* gsk = (const float*)d_in[pbase + 10];
        const float* bsk = (const float*)d_in[pbase + 11];

        // kNN graph on current points
        const ushort_t* pts = (ci == 0) ? posb : ftsCur;
        const int D = (ci == 0) ? 3 : C;
        const int pst = (ci == 0) ? 4 : Fst[ci];
        sqnorm<<<cdiv(BN, 256), 256, 0, stream>>>(pts, pst, D, sqn, BN);
        knn_kernel<<<dim3(cdiv(N, 128), B), 128, 0, stream>>>(pts, sqn, idx, N, D, pst);

        // H = [u, x_i, x_j - x_i] (bf16, K-padded)
        edge_build<<<cdiv(M * (Kp1[ci] / 8), 256), 256, 0, stream>>>(
            u_b, U, U, ftsCur, C, Fst[ci], idx, bufA, Kp1[ci], N, M * (Kp1[ci] / 8));

        // MLP stage 1
        wpack<<<cdiv((long long)O * Kp1[ci], 256), 256, 0, stream>>>(W1, wt0, U + 2 * C, O, Kp1[ci]);
        gemm_bf16<<<dim3((int)(M / 256), O / 64), 256, 0, stream>>>(bufA, wt0, bufY, (int)M, Kp1[ci], O);
        bn_stats<<<O, 256, 0, stream>>>(bufY, nullptr, M, O, mvE);
        bn_relu<<<cdiv(M * O, 256), 256, 0, stream>>>(bufY, bufA, mvE, g1, b1, M * O, O);

        // MLP stage 2
        wpack<<<cdiv((long long)O * O, 256), 256, 0, stream>>>(W2, wt1, O, O, O);
        gemm_bf16<<<dim3((int)(M / 256), O / 64), 256, 0, stream>>>(bufA, wt1, bufY, (int)M, O, O);
        bn_stats<<<O, 256, 0, stream>>>(bufY, nullptr, M, O, mvE);
        bn_relu<<<cdiv(M * O, 256), 256, 0, stream>>>(bufY, bufA, mvE, g2, b2, M * O, O);

        // MLP stage 3
        wpack<<<cdiv((long long)O * O, 256), 256, 0, stream>>>(W3, wt1, O, O, O);
        gemm_bf16<<<dim3((int)(M / 256), O / 64), 256, 0, stream>>>(bufA, wt1, bufY, (int)M, O, O);
        bn_stats<<<O, 256, 0, stream>>>(bufY, nullptr, M, O, mvE);
        bn_relu<<<cdiv(M * O, 256), 256, 0, stream>>>(bufY, bufA, mvE, g3, b3, M * O, O);

        // skip path: fts @ Ws (K = Fst padded, zero-pad cols already zeroed)
        wpack<<<cdiv((long long)O * Fst[ci], 256), 256, 0, stream>>>(Wsk, wt1, C, O, Fst[ci]);
        gemm_bf16<<<dim3((int)(BN / 256), O / 64), 256, 0, stream>>>(ftsCur, wt1, skipY, (int)BN, Fst[ci], O);
        bn_stats<<<O, 256, 0, stream>>>(skipY, nullptr, BN, O, mvS);

        // max over k edges + BN'd skip + ReLU
        aggr_skip<<<cdiv(BN * O, 256), 256, 0, stream>>>(
            bufA, skipY, mvS, gsk, bsk, ftsNext, BN * O, O);

        // u update: relu(bn([u, meanpool(fts)] @ Wgp + bgp))
        meanpool<<<cdiv((long long)B * O, 256), 256, 0, stream>>>(
            ftsNext, O, O, N, pool, (long long)B * O);
        const int gbase = 4 + (ci + 1) * 4;
        linear2<<<cdiv((long long)B * O, 256), 256, 0, stream>>>(
            u_f, U, pool, O, (const float*)d_in[gbase], (const float*)d_in[gbase + 1],
            gy, B, O, 0);
        bn_cols_relu<<<O, 128, 0, stream>>>(
            gy, (const float*)d_in[gbase + 2], (const float*)d_in[gbase + 3],
            u_f, u_b, B, O);

        ftsCur = ftsNext;
    }

    // ---- 4. classifier head ----
    linear2<<<cdiv((long long)B * 256, 256), 256, 0, stream>>>(
        u_f, 256, nullptr, 0, fcW, fcb, hbuf, B, 256, 1);
    linear2<<<1, 256, 0, stream>>>(
        hbuf, 256, nullptr, 0, outW, outb, (float*)d_out, B, 2, 0);
}
